// HybridAttention_45878840656619
// MI455X (gfx1250) — matrix operand order repaired and checked by reading, not compiled, not run on hardware
//
#include <hip/hip_runtime.h>
#include <math.h>

// ---------------------------------------------------------------------------
// Problem constants (reference: B=2, T=1024, D=1024, H=16, WIN=128, Dh=64)
// ---------------------------------------------------------------------------
#define BZ   2
#define TT   1024
#define DD   1024
#define HH   16
#define DHH  64
#define WINW 128

typedef __bf16 bf16_t;
typedef bf16_t bf16x16 __attribute__((ext_vector_type(16)));
typedef float  f32x8   __attribute__((ext_vector_type(8)));
typedef int    i32x4   __attribute__((ext_vector_type(4)));

#define DEVI __device__ __forceinline__
#define AS1 __attribute__((address_space(1)))
#define AS3 __attribute__((address_space(3)))

static DEVI float bf2f(bf16_t x) { return (float)x; }
static DEVI bf16_t f2bf(float x) { return (bf16_t)x; }

#define NEGINF (-3.0e38f)

// ---------------------------------------------------------------------------
// CDNA5 async global->LDS staging (ASYNCcnt-tracked). Guarded by __has_builtin
// so the file always compiles; falls back to a plain 16B copy otherwise.
// Builtin signature (from hipcc diagnostic): (i32x4 AS1*, i32x4 AS3*, Imm, Imm)
// ---------------------------------------------------------------------------
#if defined(__has_builtin)
#if __has_builtin(__builtin_amdgcn_global_load_async_to_lds_b128) && \
    __has_builtin(__builtin_amdgcn_s_wait_asynccnt)
#define HAVE_ASYNC_LDS 1
#endif
#endif
#ifndef HAVE_ASYNC_LDS
#define HAVE_ASYNC_LDS 0
#endif

static DEVI void cp16_g2l(void* lds_dst, const void* gsrc) {
#if HAVE_ASYNC_LDS
  __builtin_amdgcn_global_load_async_to_lds_b128(
      (AS1 i32x4*)gsrc, (AS3 i32x4*)lds_dst, 0, 0);
#else
  *(uint4*)lds_dst = *(const uint4*)gsrc;
#endif
}

static DEVI void async_wait() {
#if HAVE_ASYNC_LDS
  __builtin_amdgcn_s_wait_asynccnt(0);
#endif
}

// ---------------------------------------------------------------------------
// WMMA fragment loaders (wave32; A and B share this code's contiguous k order).
//  A (16x32, MxK): lane l: m=l&15, h=l>>4; vgpr j holds K = h*16 + 2j (+0,+1)
//  B (32x16, KxN): lane l: n=l&15, h=l>>4; vgpr j holds K = h*16 + 2j (+0,+1)
//  C/D (16x16 f32): lane l: n=l&15; vgpr r holds row m = (l>>4)*8 + r
// ---------------------------------------------------------------------------
static DEVI bf16x16 ldsA_rm(const bf16_t* s, int row0, int k0, int ld) {
  int l = threadIdx.x & 31; int m = l & 15; int hp = l >> 4;
  const bf16_t* p = s + (row0 + m) * ld + k0;
  bf16x16 f;
#pragma unroll
  for (int j = 0; j < 8; ++j) {
    int kb = (hp << 4) + (j << 1);
    f[2 * j]     = p[kb];
    f[2 * j + 1] = p[kb + 1];
  }
  return f;
}

// A stored transposed in LDS: element (m,k) = s[(k0+k)*ld + row0+m]
static DEVI bf16x16 ldsA_tr(const bf16_t* s, int row0, int k0, int ld) {
  int l = threadIdx.x & 31; int m = l & 15; int hp = l >> 4;
  bf16x16 f;
#pragma unroll
  for (int j = 0; j < 8; ++j) {
    int kb = (hp << 4) + (j << 1);
    f[2 * j]     = s[(k0 + kb) * ld + row0 + m];
    f[2 * j + 1] = s[(k0 + kb + 1) * ld + row0 + m];
  }
  return f;
}

// B stored [n][k] in LDS (contiguous in k): element (k,n) = s[(n0+n)*ld + k0+k]
static DEVI bf16x16 ldsB_nk(const bf16_t* s, int n0, int k0, int ld) {
  int l = threadIdx.x & 31; int n = l & 15; int hp = l >> 4;
  const bf16_t* p = s + (n0 + n) * ld + k0;
  bf16x16 f;
#pragma unroll
  for (int j = 0; j < 8; ++j) {
    int kb = (hp << 4) + (j << 1);
    f[2 * j]     = p[kb];
    f[2 * j + 1] = p[kb + 1];
  }
  return f;
}

// B stored [k][n] in LDS (natural row-major): element (k,n) = s[(k0+k)*ld + n0+n]
static DEVI bf16x16 ldsB_kn(const bf16_t* s, int k0, int n0, int ld) {
  int l = threadIdx.x & 31; int n = l & 15; int hp = l >> 4;
  bf16x16 f;
#pragma unroll
  for (int j = 0; j < 8; ++j) {
    int kb = (hp << 4) + (j << 1);
    f[2 * j]     = s[(k0 + kb) * ld + n0 + n];
    f[2 * j + 1] = s[(k0 + kb + 1) * ld + n0 + n];
  }
  return f;
}

// B from f32 LDS stored [k][n] (used for the linear-attention state matrix)
static DEVI bf16x16 ldsB_kn_f32(const float* s, int k0, int n0, int ld) {
  int l = threadIdx.x & 31; int n = l & 15; int hp = l >> 4;
  bf16x16 f;
#pragma unroll
  for (int j = 0; j < 8; ++j) {
    int kb = (hp << 4) + (j << 1);
    f[2 * j]     = f2bf(s[(k0 + kb) * ld + n0 + n]);
    f[2 * j + 1] = f2bf(s[(k0 + kb + 1) * ld + n0 + n]);
  }
  return f;
}

static DEVI f32x8 wmma_bf16(bf16x16 a, bf16x16 b, f32x8 c) {
  return __builtin_amdgcn_wmma_f32_16x16x32_bf16(false, a, false, b, (short)0, c,
                                                 false, false);
}

// ---------------------------------------------------------------------------
// f32 -> bf16 cast
// ---------------------------------------------------------------------------
__global__ void cast_f32_bf16(const float* __restrict__ in, bf16_t* __restrict__ out,
                              int n) {
  int i = blockIdx.x * blockDim.x + threadIdx.x;
  if (i < n) out[i] = f2bf(in[i]);
}

// ---------------------------------------------------------------------------
// GEMM: C[MxN] = epi(A[MxK] @ W[KxN]); bf16 operands, f32 accumulate.
// 256 threads = 8 waves (4x2), block tile 128x128, K step 64.
// A staged via async global->LDS; W transposed to [n][k] in LDS.
// EPI: 0 = store bf16, 1 = softplus->bf16, 2 = store f32
// ---------------------------------------------------------------------------
template <int EPI>
__global__ __launch_bounds__(256) void gemm_bf16_k(const bf16_t* __restrict__ A,
                                                   const bf16_t* __restrict__ W,
                                                   void* __restrict__ C, int M, int N,
                                                   int K) {
  __shared__ bf16_t As[128 * 64];   // [m][k]
  __shared__ bf16_t Bs[128 * 64];   // [n][k]

  const int tid = threadIdx.x;
  const int wid = tid >> 5;
  const int wm = wid >> 1, wn = wid & 1;
  const int m0 = blockIdx.y * 128;
  const int n0 = blockIdx.x * 128;

  f32x8 acc[2][4];
#pragma unroll
  for (int i = 0; i < 2; ++i)
#pragma unroll
    for (int j = 0; j < 4; ++j) acc[i][j] = (f32x8)0.0f;

  for (int k0 = 0; k0 < K; k0 += 64) {
    __syncthreads();
    // prefetch next K tile into cache (global_prefetch_b8)
    if (k0 + 64 < K) {
      __builtin_prefetch(&A[(size_t)(m0 + (tid >> 1)) * K + k0 + 64 + (tid & 1) * 32], 0, 0);
      __builtin_prefetch(&W[(size_t)(k0 + 64 + (tid >> 2)) * N + n0 + (tid & 3) * 32], 0, 0);
    }
    // A tile: 128x64, async 16B chunks (8 bf16)
#pragma unroll
    for (int it = 0; it < 4; ++it) {
      int e = tid + it * 256;          // chunk id, 1024 chunks
      int r = e >> 3, c = (e & 7) * 8;
      cp16_g2l(&As[r * 64 + c], &A[(size_t)(m0 + r) * K + k0 + c]);
    }
    // B tile: vectorized coalesced read of [k][n], transpose-store to [n][k]
#pragma unroll
    for (int it = 0; it < 4; ++it) {
      int e = tid + it * 256;          // 1024 chunks of 8
      int kk = e >> 4, nn0 = (e & 15) * 8;
      uint4 vv = *(const uint4*)&W[(size_t)(k0 + kk) * N + n0 + nn0];
      bf16_t tmp[8];
      *(uint4*)tmp = vv;
#pragma unroll
      for (int j = 0; j < 8; ++j) Bs[(nn0 + j) * 64 + kk] = tmp[j];
    }
    async_wait();
    __syncthreads();

#pragma unroll
    for (int kk = 0; kk < 64; kk += 32) {
      bf16x16 af[2], bfg[4];
#pragma unroll
      for (int i = 0; i < 2; ++i) af[i] = ldsA_rm(As, 32 * wm + 16 * i, kk, 64);
#pragma unroll
      for (int j = 0; j < 4; ++j) bfg[j] = ldsB_nk(Bs, 64 * wn + 16 * j, kk, 64);
#pragma unroll
      for (int i = 0; i < 2; ++i)
#pragma unroll
        for (int j = 0; j < 4; ++j) acc[i][j] = wmma_bf16(af[i], bfg[j], acc[i][j]);
    }
  }

  const int lane = tid & 31;
  const int nl = lane & 15, hp = lane >> 4;
#pragma unroll
  for (int i = 0; i < 2; ++i)
#pragma unroll
    for (int j = 0; j < 4; ++j)
#pragma unroll
      for (int r = 0; r < 8; ++r) {
        int row = m0 + 32 * wm + 16 * i + hp * 8 + r;
        int col = n0 + 64 * wn + 16 * j + nl;
        float v = acc[i][j][r];
        if (EPI == 1) v = (v > 20.f) ? v : log1pf(expf(v));  // softplus
        if (EPI == 2)
          ((float*)C)[(size_t)row * N + col] = v;
        else
          ((bf16_t*)C)[(size_t)row * N + col] = f2bf(v);
      }
}

// ---------------------------------------------------------------------------
// Flash attention (softmax), one WG per (b, h, 64-row tile). 128 thr = 4 waves.
// BAND=true applies the 128-wide sliding window in addition to causal.
// ---------------------------------------------------------------------------
template <bool BAND>
__global__ __launch_bounds__(128) void attn_flash_k(const bf16_t* __restrict__ q,
                                                    const bf16_t* __restrict__ k,
                                                    const bf16_t* __restrict__ v,
                                                    bf16_t* __restrict__ o) {
  __shared__ bf16_t Qs[64 * 64];
  __shared__ bf16_t Ks[64 * 64];  // [key][dh]
  __shared__ bf16_t Vs[64 * 64];  // [key][dh]
  __shared__ bf16_t Ps[64 * 64];  // [row][key]

  const int tid = threadIdx.x;
  const int wid = tid >> 5;
  const int lane = tid & 31;
  const int nl = lane & 15, hp = lane >> 4;
  const int rt = blockIdx.x, h = blockIdx.y, b = blockIdx.z;
  const int t0 = rt * 64;

  // async-stage Q (raw; Dh^-0.5 applied to scores post-WMMA)
#pragma unroll
  for (int it = 0; it < 4; ++it) {
    int e = tid + it * 128;          // 512 chunks of 8 elems
    int r = e >> 3, c = (e & 7) * 8;
    cp16_g2l(&Qs[r * 64 + c], &q[((size_t)(b * TT + t0 + r)) * DD + h * DHH + c]);
  }

  float m_r[8], l_r[8];
  f32x8 oacc[4];
#pragma unroll
  for (int r = 0; r < 8; ++r) { m_r[r] = NEGINF; l_r[r] = 0.f; }
#pragma unroll
  for (int nt = 0; nt < 4; ++nt) oacc[nt] = (f32x8)0.0f;

  int jt0 = 0;
  if (BAND) { int lo = t0 - (WINW - 1); jt0 = lo > 0 ? (lo >> 6) : 0; }

  for (int jt = jt0; jt <= rt; ++jt) {
    __syncthreads();
#pragma unroll
    for (int it = 0; it < 4; ++it) {
      int e = tid + it * 128;
      int r = e >> 3, c = (e & 7) * 8;
      size_t g = ((size_t)(b * TT + jt * 64 + r)) * DD + h * DHH + c;
      cp16_g2l(&Ks[r * 64 + c], &k[g]);
      cp16_g2l(&Vs[r * 64 + c], &v[g]);
    }
    async_wait();
    __syncthreads();

    // scores S = Q . K^T  (wave strip = 16 rows)
    f32x8 s[4];
#pragma unroll
    for (int nt = 0; nt < 4; ++nt) s[nt] = (f32x8)0.0f;
#pragma unroll
    for (int kk = 0; kk < 64; kk += 32) {
      bf16x16 af = ldsA_rm(Qs, 16 * wid, kk, 64);
#pragma unroll
      for (int nt = 0; nt < 4; ++nt) {
        bf16x16 bfg = ldsB_nk(Ks, nt * 16, kk, 64);
        s[nt] = wmma_bf16(af, bfg, s[nt]);
      }
    }

    // scale + mask
#pragma unroll
    for (int nt = 0; nt < 4; ++nt)
#pragma unroll
      for (int r = 0; r < 8; ++r) {
        int rowg = t0 + 16 * wid + hp * 8 + r;
        int colg = jt * 64 + nt * 16 + nl;
        bool ok = (colg <= rowg) && (!BAND || (rowg - colg) < WINW);
        s[nt][r] = ok ? s[nt][r] * 0.125f : NEGINF;
      }

    // online softmax
#pragma unroll
    for (int r = 0; r < 8; ++r) {
      float mx = NEGINF;
#pragma unroll
      for (int nt = 0; nt < 4; ++nt) mx = fmaxf(mx, s[nt][r]);
      mx = fmaxf(mx, __shfl_xor(mx, 1));
      mx = fmaxf(mx, __shfl_xor(mx, 2));
      mx = fmaxf(mx, __shfl_xor(mx, 4));
      mx = fmaxf(mx, __shfl_xor(mx, 8));
      float mnew = fmaxf(m_r[r], mx);
      float alpha = (m_r[r] <= NEGINF * 0.5f) ? 0.f : expf(m_r[r] - mnew);
      float rs = 0.f;
#pragma unroll
      for (int nt = 0; nt < 4; ++nt) {
        float pv = (s[nt][r] <= NEGINF * 0.5f) ? 0.f : expf(s[nt][r] - mnew);
        Ps[(16 * wid + hp * 8 + r) * 64 + nt * 16 + nl] = f2bf(pv);
        rs += pv;
      }
      rs += __shfl_xor(rs, 1);
      rs += __shfl_xor(rs, 2);
      rs += __shfl_xor(rs, 4);
      rs += __shfl_xor(rs, 8);
      l_r[r] = l_r[r] * alpha + rs;
      m_r[r] = mnew;
#pragma unroll
      for (int nt = 0; nt < 4; ++nt) oacc[nt][r] *= alpha;
    }

    // O += P . V   (same-wave LDS producer/consumer; DS ops stay in order)
#pragma unroll
    for (int kk = 0; kk < 64; kk += 32) {
      bf16x16 af = ldsA_rm(Ps, 16 * wid, kk, 64);
#pragma unroll
      for (int nt = 0; nt < 4; ++nt) {
        bf16x16 bfg = ldsB_kn(Vs, kk, nt * 16, 64);
        oacc[nt] = wmma_bf16(af, bfg, oacc[nt]);
      }
    }
  }

  // write O / l
#pragma unroll
  for (int nt = 0; nt < 4; ++nt)
#pragma unroll
    for (int r = 0; r < 8; ++r) {
      int row = t0 + 16 * wid + hp * 8 + r;
      int dh = nt * 16 + nl;
      o[((size_t)(b * TT + row)) * DD + h * DHH + dh] =
          f2bf(oacc[nt][r] / l_r[r]);
    }
}

// ---------------------------------------------------------------------------
// Chunked causal linear attention, one WG per (b,h). 128 thr = 4 waves.
// State S[64x64] f32, z[64] f32 carried across 16 chunks of 64 steps.
// ---------------------------------------------------------------------------
__global__ __launch_bounds__(128) void linear_attn_k(const bf16_t* __restrict__ qp,
                                                     const bf16_t* __restrict__ kp,
                                                     const bf16_t* __restrict__ vp,
                                                     bf16_t* __restrict__ o) {
  __shared__ bf16_t Qs[64 * 64];   // qp chunk [t][d]
  __shared__ bf16_t Kp[64 * 64];   // kp chunk [t][d]
  __shared__ bf16_t Vsh[64 * 64];  // v  chunk [t][d]
  __shared__ bf16_t Ps[64 * 64];   // masked scores [t][t']
  __shared__ float Sst[64 * 64];   // state [dh_in][dh_out]
  __shared__ float zst[64];
  __shared__ float dini[64];       // den_inter per row

  const int tid = threadIdx.x;
  const int wid = tid >> 5;
  const int lane = tid & 31;
  const int nl = lane & 15, hp = lane >> 4;
  const int h = blockIdx.x, b = blockIdx.y;

  for (int i = tid; i < 4096; i += 128) Sst[i] = 0.f;
  if (tid < 64) zst[tid] = 0.f;
  __syncthreads();

  for (int c = 0; c < TT / 64; ++c) {
    const int t0 = c * 64;
#pragma unroll
    for (int it = 0; it < 4; ++it) {
      int e = tid + it * 128;
      int r = e >> 3, d = (e & 7) * 8;
      size_t g = ((size_t)(b * TT + t0 + r)) * DD + h * DHH + d;
      cp16_g2l(&Qs[r * 64 + d], &qp[g]);
      cp16_g2l(&Kp[r * 64 + d], &kp[g]);
      cp16_g2l(&Vsh[r * 64 + d], &vp[g]);
    }
    async_wait();
    __syncthreads();

    // den_inter[t] = qp[t] . z_prev
    if (tid < 64) {
      float s = 0.f;
      for (int d = 0; d < 64; ++d) s += bf2f(Qs[tid * 64 + d]) * zst[d];
      dini[tid] = s;
    }
    __syncthreads();

    // scores A = qp . kp^T, mask t >= t'
    f32x8 s[4];
#pragma unroll
    for (int nt = 0; nt < 4; ++nt) s[nt] = (f32x8)0.0f;
#pragma unroll
    for (int kk = 0; kk < 64; kk += 32) {
      bf16x16 af = ldsA_rm(Qs, 16 * wid, kk, 64);
#pragma unroll
      for (int nt = 0; nt < 4; ++nt) {
        bf16x16 bfg = ldsB_nk(Kp, nt * 16, kk, 64);
        s[nt] = wmma_bf16(af, bfg, s[nt]);
      }
    }

    float den[8];
#pragma unroll
    for (int r = 0; r < 8; ++r) {
      int rloc = 16 * wid + hp * 8 + r;
      float rs = 0.f;
#pragma unroll
      for (int nt = 0; nt < 4; ++nt) {
        int cloc = nt * 16 + nl;
        float pv = (cloc <= rloc) ? s[nt][r] : 0.f;
        Ps[rloc * 64 + nt * 16 + nl] = f2bf(pv);
        rs += pv;
      }
      rs += __shfl_xor(rs, 1);
      rs += __shfl_xor(rs, 2);
      rs += __shfl_xor(rs, 4);
      rs += __shfl_xor(rs, 8);
      den[r] = dini[rloc] + rs + 1e-6f;
    }

    // num = qp . S_prev  +  P . v
    f32x8 oacc[4];
#pragma unroll
    for (int nt = 0; nt < 4; ++nt) oacc[nt] = (f32x8)0.0f;
#pragma unroll
    for (int kk = 0; kk < 64; kk += 32) {
      bf16x16 af = ldsA_rm(Qs, 16 * wid, kk, 64);
#pragma unroll
      for (int nt = 0; nt < 4; ++nt) {
        bf16x16 bfg = ldsB_kn_f32(Sst, kk, nt * 16, 64);
        oacc[nt] = wmma_bf16(af, bfg, oacc[nt]);
      }
    }
#pragma unroll
    for (int kk = 0; kk < 64; kk += 32) {
      bf16x16 af = ldsA_rm(Ps, 16 * wid, kk, 64);
#pragma unroll
      for (int nt = 0; nt < 4; ++nt) {
        bf16x16 bfg = ldsB_kn(Vsh, kk, nt * 16, 64);
        oacc[nt] = wmma_bf16(af, bfg, oacc[nt]);
      }
    }

    // out = num / den
#pragma unroll
    for (int nt = 0; nt < 4; ++nt)
#pragma unroll
      for (int r = 0; r < 8; ++r) {
        int row = t0 + 16 * wid + hp * 8 + r;
        int dh = nt * 16 + nl;
        o[((size_t)(b * TT + row)) * DD + h * DHH + dh] =
            f2bf(oacc[nt][r] / den[r]);
      }

    __syncthreads();  // all reads of S_prev / z_prev done

    // state update: S += kp^T . v ; z += colsum(kp)
    f32x8 su[4];
#pragma unroll
    for (int nt = 0; nt < 4; ++nt) su[nt] = (f32x8)0.0f;
#pragma unroll
    for (int kk = 0; kk < 64; kk += 32) {
      bf16x16 af = ldsA_tr(Kp, 16 * wid, kk, 64);  // (kp^T)[dh][t]
#pragma unroll
      for (int nt = 0; nt < 4; ++nt) {
        bf16x16 bfg = ldsB_kn(Vsh, kk, nt * 16, 64);
        su[nt] = wmma_bf16(af, bfg, su[nt]);
      }
    }
#pragma unroll
    for (int nt = 0; nt < 4; ++nt)
#pragma unroll
      for (int r = 0; r < 8; ++r)
        Sst[(16 * wid + hp * 8 + r) * 64 + nt * 16 + nl] += su[nt][r];
    if (tid < 64) {
      float s = zst[tid];
      for (int t = 0; t < 64; ++t) s += bf2f(Kp[t * 64 + tid]);
      zst[tid] = s;
    }
    __syncthreads();
  }
}

// ---------------------------------------------------------------------------
// Per-head gate mixture: y = sum_k softmax(gate_logits[h])[k] * proj_k
// ---------------------------------------------------------------------------
__global__ void gate_mix_k(const bf16_t* __restrict__ pl, const bf16_t* __restrict__ ps,
                           const bf16_t* __restrict__ pf,
                           const float* __restrict__ gl, bf16_t* __restrict__ y,
                           int n) {
  int i = blockIdx.x * blockDim.x + threadIdx.x;
  if (i >= n) return;
  int d = i & (DD - 1);
  int h = d >> 6;
  float g0 = gl[h * 3 + 0], g1 = gl[h * 3 + 1], g2 = gl[h * 3 + 2];
  float m = fmaxf(g0, fmaxf(g1, g2));
  float e0 = expf(g0 - m), e1 = expf(g1 - m), e2 = expf(g2 - m);
  float inv = 1.f / (e0 + e1 + e2);
  float v = (e0 * bf2f(pl[i]) + e1 * bf2f(ps[i]) + e2 * bf2f(pf[i])) * inv;
  y[i] = f2bf(v);
}

// ---------------------------------------------------------------------------
// Host orchestration
// ---------------------------------------------------------------------------
extern "C" void kernel_launch(void* const* d_in, const int* in_sizes, int n_in,
                              void* d_out, int out_size, void* d_ws, size_t ws_size,
                              hipStream_t stream) {
  const float* x = (const float*)d_in[0];
  const float* Wf[13];
  for (int i = 0; i < 13; ++i) Wf[i] = (const float*)d_in[1 + i];
  const float* gl = (const float*)d_in[14];
  float* out = (float*)d_out;

  const size_t E = (size_t)BZ * TT * DD;      // 2,097,152
  const size_t EW = (size_t)DD * DD;          // 1,048,576

  char* p = (char*)d_ws;
  auto alloc = [&](size_t bytes) -> void* {
    void* r = (void*)p;
    p += (bytes + 255) & ~(size_t)255;
    return r;
  };

  bf16_t* xbf = (bf16_t*)alloc(E * 2);
  bf16_t* Wb[13];
  for (int i = 0; i < 13; ++i) Wb[i] = (bf16_t*)alloc(EW * 2);
  bf16_t* qL = (bf16_t*)alloc(E * 2); bf16_t* kL = (bf16_t*)alloc(E * 2);
  bf16_t* vL = (bf16_t*)alloc(E * 2);
  bf16_t* qS = (bf16_t*)alloc(E * 2); bf16_t* kS = (bf16_t*)alloc(E * 2);
  bf16_t* vS = (bf16_t*)alloc(E * 2);
  bf16_t* qF = (bf16_t*)alloc(E * 2); bf16_t* kF = (bf16_t*)alloc(E * 2);
  bf16_t* vF = (bf16_t*)alloc(E * 2);
  bf16_t* aL = (bf16_t*)alloc(E * 2); bf16_t* aS = (bf16_t*)alloc(E * 2);
  bf16_t* aF = (bf16_t*)alloc(E * 2);
  bf16_t* pL = (bf16_t*)alloc(E * 2); bf16_t* pS = (bf16_t*)alloc(E * 2);
  bf16_t* pF = (bf16_t*)alloc(E * 2);
  bf16_t* ym = (bf16_t*)alloc(E * 2);

  // casts
  cast_f32_bf16<<<(int)((E + 255) / 256), 256, 0, stream>>>(x, xbf, (int)E);
  for (int i = 0; i < 13; ++i)
    cast_f32_bf16<<<(int)((EW + 255) / 256), 256, 0, stream>>>(Wf[i], Wb[i], (int)EW);

  const int M = BZ * TT, N = DD, K = DD;
  dim3 gg(N / 128, M / 128);
  auto gemm = [&](const bf16_t* A, const bf16_t* W, void* C, int epi) {
    if (epi == 0)
      gemm_bf16_k<0><<<gg, 256, 0, stream>>>(A, W, C, M, N, K);
    else if (epi == 1)
      gemm_bf16_k<1><<<gg, 256, 0, stream>>>(A, W, C, M, N, K);
    else
      gemm_bf16_k<2><<<gg, 256, 0, stream>>>(A, W, C, M, N, K);
  };

  // projections (linear branch q,k get softplus epilogue)
  gemm(xbf, Wb[0], qL, 1);  // Wlq
  gemm(xbf, Wb[1], kL, 1);  // Wlk
  gemm(xbf, Wb[2], vL, 0);  // Wlv
  gemm(xbf, Wb[4], qS, 0);  // Wsq
  gemm(xbf, Wb[5], kS, 0);  // Wsk
  gemm(xbf, Wb[6], vS, 0);  // Wsv
  gemm(xbf, Wb[8], qF, 0);  // Wfq
  gemm(xbf, Wb[9], kF, 0);  // Wfk
  gemm(xbf, Wb[10], vF, 0); // Wfv

  // attention branches
  linear_attn_k<<<dim3(HH, BZ), 128, 0, stream>>>(qL, kL, vL, aL);
  attn_flash_k<true><<<dim3(TT / 64, HH, BZ), 128, 0, stream>>>(qS, kS, vS, aS);
  attn_flash_k<false><<<dim3(TT / 64, HH, BZ), 128, 0, stream>>>(qF, kF, vF, aF);

  // output projections per branch
  gemm(aL, Wb[3], pL, 0);   // Wlo
  gemm(aS, Wb[7], pS, 0);   // Wso
  gemm(aF, Wb[11], pF, 0);  // Wfo

  // per-head gated mixture
  gate_mix_k<<<(int)((E + 255) / 256), 256, 0, stream>>>(pL, pS, pF, gl, ym, (int)E);

  // final projection -> f32 output
  gemm(ym, Wb[12], out, 2);  // Wout
}
